// GNN_19868518711955
// MI455X (gfx1250) — matrix operand order repaired and checked by reading, not compiled, not run on hardware
//
#include <hip/hip_runtime.h>
#include <hip/hip_bf16.h>

// ---------------------------------------------------------------------------
// GNN on MI455X (gfx1250): message MLP factored to per-node, WMMA f16 GEMMs
// with pre-shuffled weight fragments, f32 atomics for edge aggregation /
// batchnorm stats / pooling.
// ---------------------------------------------------------------------------

typedef __attribute__((ext_vector_type(16))) _Float16 v16h;
typedef __attribute__((ext_vector_type(8)))  float    v8f;

#define DD   64
#define LAY  4
#define EPSn 1e-5f

__device__ __forceinline__ v8f wmma_f16(v16h a, v16h b, v8f c) {
    // emits v_wmma_f32_16x16x32_f16
    return __builtin_amdgcn_wmma_f32_16x16x32_f16(
        /*neg_a=*/false, a, /*neg_b=*/false, b,
        /*c_mod=*/(short)0, c, /*reuse_a=*/false, /*reuse_b=*/false);
}

// ---------------- weight shuffle: f32 row-major -> f16 B-fragment layout ---
// Fragment (layer li, column-tile wv, k-tile k): lane holds 16 contiguous
// halfs.  dst[li*4096 + ((wv*2+k)*32 + lane)*16 + j] =
//   src[li*4096 + (k*32 + (lane>>4)*8 + (j&7) + (j>=8?16:0))*64 + wv*16 + (lane&15)]
// With tid = li*4096 + wv*1024 + k*512 + lane*16 + j, dst index == tid.
__global__ void shuffle_w_kernel(const float* __restrict__ src,
                                 _Float16* __restrict__ dst, int total) {
    int tid = blockIdx.x * blockDim.x + threadIdx.x;
    if (tid >= total) return;
    int j    = tid & 15;
    int lane = (tid >> 4) & 31;
    int k    = (tid >> 9) & 1;
    int wv   = (tid >> 10) & 3;
    int li   = tid >> 12;
    int hs = lane >> 4, n16 = lane & 15;
    int kk  = k * 32 + hs * 8 + (j & 7) + ((j >> 3) << 4);
    int col = wv * 16 + n16;
    dst[tid] = (_Float16)src[li * 4096 + kk * DD + col];
}

__global__ void zero_kernel(float* __restrict__ p, int n) {
    int i = blockIdx.x * blockDim.x + threadIdx.x;
    if (i < n) p[i] = 0.f;
}

// ---------------- per-node MLP + root transform (WMMA, persistent) ---------
// block = 128 threads (4 waves). Each iteration processes 16 node rows.
// Wave w owns output columns [w*16, w*16+16).
//   msg = relu(h@W1+b1)@W2+b2      -> msg buffer
//   agg = h@Wr+br                  -> agg buffer (edge kernel atomically adds)
__global__ __launch_bounds__(128)
void mlp_kernel(const float* __restrict__ hin,
                const _Float16* __restrict__ W1, const float* __restrict__ b1,
                const _Float16* __restrict__ W2, const float* __restrict__ b2,
                const _Float16* __restrict__ Wr, const float* __restrict__ br,
                float* __restrict__ msg, float* __restrict__ agg, int N)
{
    __shared__ _Float16 tbuf[16 * DD];   // relu(h@W1+b1) staged for transpose
    __shared__ float    outm[16 * DD];   // msg staging (coalesced store)
    __shared__ float    outa[16 * DD];   // agg staging

    const int tid  = threadIdx.x;
    const int lane = tid & 31;
    const int wv   = tid >> 5;       // column tile 0..3
    const int n16  = lane & 15;
    const int hs   = lane >> 4;      // half-wave select
    const int col  = wv * 16 + n16;  // global output column for B/C frags

    // ---- B fragments: contiguous 32B per lane (pre-shuffled layout) ----
    v16h B1[2], B2[2], BR[2];
#pragma unroll
    for (int k = 0; k < 2; ++k) {
        const long fo = ((long)(wv * 2 + k) * 32 + lane) * 16;
        B1[k] = *(const v16h*)(W1 + fo);
        B2[k] = *(const v16h*)(W2 + fo);
        BR[k] = *(const v16h*)(Wr + fo);
    }
    const float bias1 = b1[col], bias2 = b2[col], biasr = br[col];

    const int ntiles = (N + 15) >> 4;
    for (int tile = blockIdx.x; tile < ntiles; tile += gridDim.x) {
        const long rowbase = (long)tile * 16;

        // prefetch next tile's A row for this lane (global_prefetch_b8)
        const int nt = tile + gridDim.x;
        if (nt < ntiles) {
            const float* nrp = hin + ((long)nt * 16 + n16) * DD;
            __builtin_prefetch(nrp, 0, 1);
            __builtin_prefetch(nrp + 32, 0, 1);
        }

        // ---- A fragments from hin (f32 -> f16) ----
        // A 16x32 f16 layout: lane row M = lane&15; element j -> K =
        // (j&7) + (j>=8 ? 16 : 0) + hs*8  (two contiguous 8-float chunks).
        const float* rp = hin + (rowbase + n16) * DD;
        v16h A[2];
#pragma unroll
        for (int k = 0; k < 2; ++k) {
            const float* p = rp + k * 32 + hs * 8;
#pragma unroll
            for (int j = 0; j < 8; ++j) A[k][j]     = (_Float16)p[j];
#pragma unroll
            for (int j = 0; j < 8; ++j) A[k][8 + j] = (_Float16)p[16 + j];
        }

        // ---- GEMM1: relu(h@W1 + b1) ----
        v8f acc1 = {};
        acc1 = wmma_f16(A[0], B1[0], acc1);
        acc1 = wmma_f16(A[1], B1[1], acc1);
#pragma unroll
        for (int v = 0; v < 8; ++v) {
            float y = acc1[v] + bias1;
            y = y > 0.f ? y : 0.f;
            // C/D layout: row M = hs*8 + v, col = lane&15 (+tile)
            tbuf[(hs * 8 + v) * DD + col] = (_Float16)y;
        }
        __syncthreads();

        // ---- A2 fragments from LDS (cross-lane transpose via LDS) ----
        v16h A2[2];
        {
            const _Float16* tp = tbuf + n16 * DD;
#pragma unroll
            for (int k = 0; k < 2; ++k) {
                const _Float16* p = tp + k * 32 + hs * 8;
#pragma unroll
                for (int j = 0; j < 8; ++j) A2[k][j]     = p[j];
#pragma unroll
                for (int j = 0; j < 8; ++j) A2[k][8 + j] = p[16 + j];
            }
        }

        // ---- GEMM2: msg = t@W2 + b2 ;  GEMM3: root = h@Wr + br ----
        v8f acc2 = {};
        acc2 = wmma_f16(A2[0], B2[0], acc2);
        acc2 = wmma_f16(A2[1], B2[1], acc2);
        v8f acc3 = {};
        acc3 = wmma_f16(A[0], BR[0], acc3);
        acc3 = wmma_f16(A[1], BR[1], acc3);
#pragma unroll
        for (int v = 0; v < 8; ++v) {
            outm[(hs * 8 + v) * DD + col] = acc2[v] + bias2;
            outa[(hs * 8 + v) * DD + col] = acc3[v] + biasr;
        }
        __syncthreads();

        // ---- coalesced 128-bit stores: 1024 floats per buffer ----
        float4*       mg = (float4*)(msg + rowbase * DD);
        float4*       ag = (float4*)(agg + rowbase * DD);
        const float4* sm = (const float4*)outm;
        const float4* sa = (const float4*)outa;
#pragma unroll
        for (int t = tid; t < 256; t += 128) { mg[t] = sm[t]; ag[t] = sa[t]; }
    }
}

// ---------------- edge scatter: agg[dst] += msg[src] -----------------------
// One thread per (edge, 4-float chunk): E*16 threads.
__global__ __launch_bounds__(256)
void edge_kernel(const int* __restrict__ ei, const float* __restrict__ msg,
                 float* __restrict__ agg, int E)
{
    long tid = (long)blockIdx.x * blockDim.x + threadIdx.x;
    if (tid >= (long)E * 16) return;
    int e = (int)(tid >> 4);
    int c = (int)(tid & 15) * 4;
    int s = ei[e];
    int d = ei[E + e];
    const float4 v = *(const float4*)(msg + (long)s * DD + c);
    float* ap = agg + (long)d * DD + c;
    atomicAdd(ap + 0, v.x);
    atomicAdd(ap + 1, v.y);
    atomicAdd(ap + 2, v.z);
    atomicAdd(ap + 3, v.w);
}

// ---------------- batchnorm stats: per-channel sum / sumsq of prelu(agg) ---
__global__ __launch_bounds__(256)
void stats_kernel(const float* __restrict__ agg,
                  const float* __restrict__ aconv, int li,
                  float* __restrict__ sums, int N)
{
    __shared__ float s1[256], s2[256];
    const float alpha = aconv[li];
    const int tid = threadIdx.x;
    const int d   = tid & 63;
    const int r0  = tid >> 6;  // 0..3
    float sum = 0.f, sq = 0.f;
    for (int row = blockIdx.x * 4 + r0; row < N; row += gridDim.x * 4) {
        float y = agg[(long)row * DD + d];
        y = y >= 0.f ? y : alpha * y;
        sum += y; sq += y * y;
    }
    s1[tid] = sum; s2[tid] = sq;
    __syncthreads();
    if (r0 == 0) {
#pragma unroll
        for (int k = 1; k < 4; ++k) { sum += s1[d + 64 * k]; sq += s2[d + 64 * k]; }
        atomicAdd(&sums[d], sum);
        atomicAdd(&sums[64 + d], sq);
    }
}

// ---------------- fused prelu + batchnorm write-back -----------------------
__global__ __launch_bounds__(256)
void norm_kernel(const float* __restrict__ agg,
                 const float* __restrict__ sums,
                 const float* __restrict__ aconv, int li,
                 const float* __restrict__ gamma, const float* __restrict__ beta,
                 float* __restrict__ hout, int N)
{
    long tid = (long)blockIdx.x * blockDim.x + threadIdx.x;
    if (tid >= (long)N * 16) return;
    const float alpha = aconv[li];
    const float invN  = 1.f / (float)N;
    int row = (int)(tid >> 4);
    int c0  = (int)(tid & 15) * 4;
    float4 v = *(const float4*)(agg + (long)row * DD + c0);
    float in[4] = { v.x, v.y, v.z, v.w };
    float r[4];
#pragma unroll
    for (int j = 0; j < 4; ++j) {
        int c = c0 + j;
        float y  = in[j] >= 0.f ? in[j] : alpha * in[j];
        float mu = sums[c] * invN;
        float vr = sums[64 + c] * invN - mu * mu;
        r[j] = gamma[c] * (y - mu) * rsqrtf(vr + EPSn) + beta[c];
    }
    *(float4*)(hout + (long)row * DD + c0) = make_float4(r[0], r[1], r[2], r[3]);
}

// ---------------- global mean pool (atomic accumulation) -------------------
__global__ __launch_bounds__(256)
void pool_kernel(const float* __restrict__ h, const int* __restrict__ batch,
                 float* __restrict__ psum, float* __restrict__ pcnt, int N)
{
    long tid = (long)blockIdx.x * blockDim.x + threadIdx.x;
    if (tid >= (long)N * 16) return;
    int row = (int)(tid >> 4);
    int c0  = (int)(tid & 15) * 4;
    int g   = batch[row];
    float4 v = *(const float4*)(h + (long)row * DD + c0);
    float* p = psum + (long)g * DD + c0;
    atomicAdd(p + 0, v.x);
    atomicAdd(p + 1, v.y);
    atomicAdd(p + 2, v.z);
    atomicAdd(p + 3, v.w);
    if (c0 == 0) atomicAdd(&pcnt[g], 1.f);
}

// ---------------- MLP head: out[g] = prelu(r@Wh1+bh1)@Wh2 + bh2 ------------
__global__ __launch_bounds__(64)
void head_kernel(const float* __restrict__ psum, const float* __restrict__ pcnt,
                 const float* __restrict__ Wh1, const float* __restrict__ bh1,
                 const float* __restrict__ Wh2, const float* __restrict__ bh2,
                 const float* __restrict__ ahead, float* __restrict__ out)
{
    __shared__ float r[DD];
    __shared__ float hdn[DD];
    const int g = blockIdx.x;
    const int d = threadIdx.x;
    float cnt = pcnt[g]; cnt = cnt > 1.f ? cnt : 1.f;
    r[d] = psum[(long)g * DD + d] / cnt;
    __syncthreads();
    float acc = bh1[d];
#pragma unroll 8
    for (int k = 0; k < DD; ++k) acc += r[k] * Wh1[k * DD + d];
    const float a = ahead[0];
    acc = acc >= 0.f ? acc : a * acc;
    hdn[d] = acc * Wh2[d];   // OUT == 1
    __syncthreads();
    if (d == 0) {
        float s = bh2[0];
        for (int k = 0; k < DD; ++k) s += hdn[k];
        out[g] = s;
    }
}

// ---------------------------------------------------------------------------
extern "C" void kernel_launch(void* const* d_in, const int* in_sizes, int n_in,
                              void* d_out, int out_size, void* d_ws, size_t ws_size,
                              hipStream_t stream)
{
    const float* x     = (const float*)d_in[0];
    const int*   ei    = (const int*)  d_in[1];
    const int*   batch = (const int*)  d_in[2];
    const float* Wm1   = (const float*)d_in[3];
    const float* bm1   = (const float*)d_in[4];
    const float* Wm2   = (const float*)d_in[5];
    const float* bm2   = (const float*)d_in[6];
    const float* Wr    = (const float*)d_in[7];
    const float* br    = (const float*)d_in[8];
    const float* aconv = (const float*)d_in[9];
    const float* gamma = (const float*)d_in[10];
    const float* beta  = (const float*)d_in[11];
    const float* Wh1   = (const float*)d_in[12];
    const float* bh1   = (const float*)d_in[13];
    const float* Wh2   = (const float*)d_in[14];
    const float* bh2   = (const float*)d_in[15];
    const float* ahead = (const float*)d_in[16];
    float* out = (float*)d_out;

    const int N = in_sizes[0] / DD;      // 50000
    const int E = in_sizes[1] / 2;       // 800000
    const int G = out_size;              // 64 (OUT == 1)

    // ---- workspace layout ----
    float* H    = (float*)d_ws;          // N*64
    float* MSG  = H   + (long)N * DD;    // N*64
    float* AGG  = MSG + (long)N * DD;    // N*64
    float* SUMS = AGG + (long)N * DD;    // 128
    float* PS   = SUMS + 128;            // G*64
    float* PC   = PS + (long)G * DD;     // G
    // f16 weight fragments, 32B aligned for v16h loads
    long foff = ((long)(PC + G - (float*)d_ws) + 15) & ~15L;
    _Float16* W1h = (_Float16*)((float*)d_ws + foff);   // L*64*64
    _Float16* W2h = W1h + LAY * DD * DD;
    _Float16* WRh = W2h + LAY * DD * DD;

    const int nW = LAY * DD * DD;        // 16384 per weight stack
    shuffle_w_kernel<<<(nW + 255) / 256, 256, 0, stream>>>(Wm1, W1h, nW);
    shuffle_w_kernel<<<(nW + 255) / 256, 256, 0, stream>>>(Wm2, W2h, nW);
    shuffle_w_kernel<<<(nW + 255) / 256, 256, 0, stream>>>(Wr,  WRh, nW);

    const int ntiles    = (N + 15) / 16;                 // 3125 (N % 16 == 0)
    const int mlpBlocks = ntiles < 1024 ? ntiles : 1024; // persistent grid
    const long eThreads = (long)E * 16;
    const int  eBlocks  = (int)((eThreads + 255) / 256);
    const long nThreads = (long)N * 16;
    const int  nBlocks  = (int)((nThreads + 255) / 256);

    for (int li = 0; li < LAY; ++li) {
        const float* hin = (li == 0) ? x : H;
        zero_kernel<<<1, 128, 0, stream>>>(SUMS, 128);
        mlp_kernel<<<mlpBlocks, 128, 0, stream>>>(
            hin,
            W1h + li * DD * DD, bm1 + li * DD,
            W2h + li * DD * DD, bm2 + li * DD,
            WRh + li * DD * DD, br  + li * DD,
            MSG, AGG, N);
        edge_kernel<<<eBlocks, 256, 0, stream>>>(ei, MSG, AGG, E);
        stats_kernel<<<512, 256, 0, stream>>>(AGG, aconv, li, SUMS, N);
        norm_kernel<<<nBlocks, 256, 0, stream>>>(
            AGG, SUMS, aconv, li, gamma + li * DD, beta + li * DD, H, N);
    }

    zero_kernel<<<(G * DD + G + 127) / 128, 128, 0, stream>>>(PS, G * DD + G);
    pool_kernel<<<nBlocks, 256, 0, stream>>>(H, batch, PS, PC, N);
    head_kernel<<<G, DD, 0, stream>>>(PS, PC, Wh1, bh1, Wh2, bh2, ahead, out);
}
